// Block_71554155151873
// MI455X (gfx1250) — matrix operand order repaired and checked by reading, not compiled, not run on hardware
//
#include <hip/hip_runtime.h>

#define D_    64
#define S_    16
#define H_    8
#define R_    64
#define DH_   512
#define KPRE_ 192

typedef __attribute__((ext_vector_type(16))) __bf16        v16bf;
typedef __attribute__((ext_vector_type(8)))  float         v8f;
typedef __attribute__((ext_vector_type(4)))  unsigned int  u32x4;
typedef unsigned short bf16u;

__device__ __forceinline__ bf16u f2bf(float f) {
  unsigned u = __float_as_uint(f);
  u += 0x7FFFu + ((u >> 16) & 1u);   // round-to-nearest-even
  return (bf16u)(u >> 16);
}
__device__ __forceinline__ float bf2f(bf16u h) {
  return __uint_as_float(((unsigned)h) << 16);
}
__device__ __forceinline__ float siluf(float z) {
  return z * (1.0f / (1.0f + __expf(-z)));
}

union FragU { v16bf v; u32x4 q[2]; unsigned short h[16]; };

// A fragment: row-major [16 x >=32] bf16 tile, row stride lda (elements).
// ISA layout: lane<16 -> row=lane, K {0..7,16..23}; lane>=16 -> row=lane-16, K {8..15,24..31}
__device__ __forceinline__ v16bf fragA(const bf16u* p0, int lda) {
  int lane = threadIdx.x & 31;
  const bf16u* p = p0 + (lane & 15) * lda + ((lane >> 4) << 3);
  FragU r;
  r.q[0] = *(const u32x4*)(p);
  r.q[1] = *(const u32x4*)(p + 16);
  return r.v;
}
// B fragment: K-major (transposed) weights, col stride ldk, K window [k0,k0+32).
// ISA layout: col=lane&15; lane<16 -> K {k0..k0+7,k0+16..k0+23}; lane>=16 -> K {k0+8..k0+15,k0+24..k0+31}
__device__ __forceinline__ v16bf fragB(const bf16u* bt, int ldk, int k0) {
  int lane = threadIdx.x & 31;
  const bf16u* p = bt + (lane & 15) * ldk + k0 + ((lane >> 4) << 3);
  FragU r;
  r.q[0] = *(const u32x4*)(p);
  r.q[1] = *(const u32x4*)(p + 16);
  return r.v;
}
__device__ __forceinline__ v8f wmma_bf16(v16bf a, v16bf b, v8f c) {
  return __builtin_amdgcn_wmma_f32_16x16x32_bf16(false, a, false, b, (short)0, c,
                                                 false, false);
}

// ---------------- weight prep ----------------
// dst[n*K + k] = bf16(src[k*N + n])   (transpose to K-major for B fragments)
__global__ void kConvT(const float* __restrict__ src, bf16u* __restrict__ dst,
                       int K, int N) {
  int total = K * N;
  for (int idx = blockIdx.x * blockDim.x + threadIdx.x; idx < total;
       idx += gridDim.x * blockDim.x) {
    int n = idx / K;
    int k = idx - n * K;
    dst[idx] = f2bf(src[(size_t)k * N + n]);
  }
}
// same, but pad N up to Npad columns of zeros (for a3: 8 -> 16)
__global__ void kConvTpad(const float* __restrict__ src, bf16u* __restrict__ dst,
                          int K, int N, int Npad) {
  int total = K * Npad;
  for (int idx = blockIdx.x * blockDim.x + threadIdx.x; idx < total;
       idx += gridDim.x * blockDim.x) {
    int n = idx / K;
    int k = idx - n * K;
    dst[idx] = (n < N) ? f2bf(src[(size_t)k * N + n]) : (bf16u)0;
  }
}
__global__ void kFillF(float* __restrict__ p, int n, float v) {
  for (int i = blockIdx.x * blockDim.x + threadIdx.x; i < n;
       i += gridDim.x * blockDim.x)
    p[i] = v;
}

// ---------------- pre_lin: msg = [fea[src]|fea[dst]|elen] @ W_pre + b_pre ----------------
// block = 256 thr = 8 waves; wave -> one 16-edge tile; block -> 128 edges
__global__ __launch_bounds__(256) void kMsg(
    const float* __restrict__ node_fea, const float* __restrict__ elen,
    const int* __restrict__ esrc, const int* __restrict__ edst,
    const bf16u* __restrict__ WpreT, const float* __restrict__ b_pre,
    bf16u* __restrict__ msgw) {
  __shared__ __align__(16) bf16u Als[128][KPRE_];
  int tid = threadIdx.x;
  int e0 = blockIdx.x * 128;
  for (int idx = tid; idx < 128 * 64; idx += 256) {
    int er = idx >> 6, c = idx & 63;
    int e = e0 + er;
    int s = esrc[e], d = edst[e];
    Als[er][c]       = f2bf(node_fea[(size_t)s * 64 + c]);
    Als[er][64 + c]  = f2bf(node_fea[(size_t)d * 64 + c]);
    Als[er][128 + c] = f2bf(elen[(size_t)e * 64 + c]);
  }
  __syncthreads();
  int wave = tid >> 5, lane = tid & 31;
  int wr = wave * 16;
  int rowbase = (lane >> 4) * 8, col0 = lane & 15;
  v8f c[4] = {};
  for (int kb = 0; kb < 6; ++kb) {
    v16bf a = fragA(&Als[wr][kb * 32], KPRE_);
#pragma unroll
    for (int nt = 0; nt < 4; ++nt) {
      v16bf b = fragB(WpreT + nt * 16 * KPRE_, KPRE_, kb * 32);
      c[nt] = wmma_bf16(a, b, c[nt]);
    }
  }
#pragma unroll
  for (int nt = 0; nt < 4; ++nt) {
    int col = nt * 16 + col0;
    float bp = b_pre[col];
#pragma unroll
    for (int i = 0; i < 8; ++i) {
      int e = e0 + wr + rowbase + i;
      msgw[(size_t)e * 64 + col] = f2bf(c[nt][i] + bp);
    }
  }
}

// ---------------- attention logits: 2x (Linear+LN+SiLU) + Linear, atomic max ----------------
__global__ __launch_bounds__(256) void kAlpha(
    const float* __restrict__ elen, const int* __restrict__ edst,
    const bf16u* __restrict__ a1T, const float* __restrict__ ab1,
    const float* __restrict__ g1, const float* __restrict__ be1,
    const bf16u* __restrict__ a2T, const float* __restrict__ ab2,
    const float* __restrict__ g2, const float* __restrict__ be2,
    const bf16u* __restrict__ a3T, const float* __restrict__ ab3,
    float* __restrict__ alpha_out, float* __restrict__ mbuf) {
  __shared__ __align__(16) bf16u Xb[64][64];
  __shared__ float Xf[64][64];
  int tid = threadIdx.x;
  int e0 = blockIdx.x * 64;
  for (int idx = tid; idx < 64 * 64; idx += 256) {
    int r = idx >> 6, cc = idx & 63;
    Xb[r][cc] = f2bf(elen[(size_t)(e0 + r) * 64 + cc]);
  }
  __syncthreads();
  int wave = tid >> 5, lane = tid & 31;
  int rowbase = (lane >> 4) * 8, col0 = lane & 15;

#pragma unroll
  for (int t = 0; t < 2; ++t) {
    int tile = wave * 2 + t;
    int rt = tile >> 2, ct = tile & 3;
    v8f acc = {};
    acc = wmma_bf16(fragA(&Xb[rt * 16][0], 64), fragB(a1T + ct * 16 * 64, 64, 0), acc);
    acc = wmma_bf16(fragA(&Xb[rt * 16][32], 64), fragB(a1T + ct * 16 * 64, 64, 32), acc);
    int cg = ct * 16 + col0;
    float ab = ab1[cg];
#pragma unroll
    for (int i = 0; i < 8; ++i) Xf[rt * 16 + rowbase + i][cg] = acc[i] + ab;
  }
  __syncthreads();
  {
    int r = tid >> 2, sub = tid & 3;
    float s = 0.f;
#pragma unroll
    for (int j = 0; j < 16; ++j) s += Xf[r][sub * 16 + j];
    s += __shfl_xor(s, 1);
    s += __shfl_xor(s, 2);
    float mu = s * (1.0f / 64.0f);
    float v = 0.f;
#pragma unroll
    for (int j = 0; j < 16; ++j) { float d = Xf[r][sub * 16 + j] - mu; v += d * d; }
    v += __shfl_xor(v, 1);
    v += __shfl_xor(v, 2);
    float inv = rsqrtf(v * (1.0f / 64.0f) + 1e-6f);
#pragma unroll
    for (int j = 0; j < 16; ++j) {
      int cc = sub * 16 + j;
      float z = (Xf[r][cc] - mu) * inv * g1[cc] + be1[cc];
      Xb[r][cc] = f2bf(siluf(z));
    }
  }
  __syncthreads();
#pragma unroll
  for (int t = 0; t < 2; ++t) {
    int tile = wave * 2 + t;
    int rt = tile >> 2, ct = tile & 3;
    v8f acc = {};
    acc = wmma_bf16(fragA(&Xb[rt * 16][0], 64), fragB(a2T + ct * 16 * 64, 64, 0), acc);
    acc = wmma_bf16(fragA(&Xb[rt * 16][32], 64), fragB(a2T + ct * 16 * 64, 64, 32), acc);
    int cg = ct * 16 + col0;
    float ab = ab2[cg];
#pragma unroll
    for (int i = 0; i < 8; ++i) Xf[rt * 16 + rowbase + i][cg] = acc[i] + ab;
  }
  __syncthreads();
  {
    int r = tid >> 2, sub = tid & 3;
    float s = 0.f;
#pragma unroll
    for (int j = 0; j < 16; ++j) s += Xf[r][sub * 16 + j];
    s += __shfl_xor(s, 1);
    s += __shfl_xor(s, 2);
    float mu = s * (1.0f / 64.0f);
    float v = 0.f;
#pragma unroll
    for (int j = 0; j < 16; ++j) { float d = Xf[r][sub * 16 + j] - mu; v += d * d; }
    v += __shfl_xor(v, 1);
    v += __shfl_xor(v, 2);
    float inv = rsqrtf(v * (1.0f / 64.0f) + 1e-6f);
#pragma unroll
    for (int j = 0; j < 16; ++j) {
      int cc = sub * 16 + j;
      float z = (Xf[r][cc] - mu) * inv * g2[cc] + be2[cc];
      Xb[r][cc] = f2bf(siluf(z));
    }
  }
  __syncthreads();
  if (wave < 4) {
    int rt = wave;
    v8f acc = {};
    acc = wmma_bf16(fragA(&Xb[rt * 16][0], 64), fragB(a3T, 64, 0), acc);
    acc = wmma_bf16(fragA(&Xb[rt * 16][32], 64), fragB(a3T, 64, 32), acc);
    if (col0 < 8) {
      float ab = ab3[col0];
#pragma unroll
      for (int i = 0; i < 8; ++i) {
        int e = e0 + rt * 16 + rowbase + i;
        float va = acc[i] + ab;
        alpha_out[(size_t)e * 8 + col0] = va;
        float* addr = &mbuf[(size_t)edst[e] * 8 + col0];
        if (va >= 0.f) atomicMax((int*)addr, __float_as_int(va));
        else           atomicMin((unsigned int*)addr, __float_as_uint(va));
      }
    }
  }
}

// ---------------- softmax pass 2: ea = exp(a - m), denom += ea ----------------
__global__ void kEa(const int* __restrict__ edst, const float* __restrict__ mbuf,
                    float* __restrict__ alpha_io, float* __restrict__ denom, int E) {
  int e = blockIdx.x * blockDim.x + threadIdx.x;
  if (e >= E) return;
  int d = edst[e];
#pragma unroll
  for (int h = 0; h < 8; ++h) {
    float a = alpha_io[(size_t)e * 8 + h];
    float m = mbuf[(size_t)d * 8 + h];
    float ea = __expf(a - m);
    alpha_io[(size_t)e * 8 + h] = ea;
    atomicAdd(&denom[(size_t)d * 8 + h], ea);
  }
}

// ---------------- main: tp2 GEMM + radial gate + attention weight + scatter ----------------
// block = 256 thr = 8 waves; 64 edges/block; wave -> 32 edges x 128 cols.
// A fragments built per-lane in registers from the rank-1 structure xs = msg (x) sh:
//   for K window [32kb,32kb+32): lane needs msg[row][2kb], msg[row][2kb+1] (one dword)
//   and sh[row][sb..sb+7] (loop-invariant). -> barrier-free inner loop.
__global__ __launch_bounds__(256) void kMain(
    const bf16u* __restrict__ msgw, const float* __restrict__ edge_sh,
    const float* __restrict__ elen, const int* __restrict__ edst,
    const bf16u* __restrict__ r1T, const float* __restrict__ rb1,
    const bf16u* __restrict__ r2T, const float* __restrict__ rb2,
    const bf16u* __restrict__ Wtp2T,
    const float* __restrict__ ea, const float* __restrict__ denom,
    float* __restrict__ nagg) {
  __shared__ __align__(16) bf16u msgT[64][64];
  __shared__ __align__(16) bf16u shT[64][16];
  __shared__ __align__(16) bf16u Eb[64][64];
  __shared__ __align__(16) bf16u tG[64][64];
  __shared__ float wfac[64][8];
  __shared__ int dstL[64];

  int tid = threadIdx.x;
  int e0 = blockIdx.x * 64;
  if (tid < 64) dstL[tid] = edst[e0 + tid];
  for (int idx = tid; idx < 64 * 64; idx += 256) {
    int r = idx >> 6, cc = idx & 63;
    msgT[r][cc] = msgw[(size_t)(e0 + r) * 64 + cc];
    Eb[r][cc]   = f2bf(elen[(size_t)(e0 + r) * 64 + cc]);
  }
  for (int idx = tid; idx < 64 * 16; idx += 256) {
    int r = idx >> 4, cc = idx & 15;
    shT[r][cc] = f2bf(edge_sh[(size_t)(e0 + r) * 16 + cc]);
  }
  __syncthreads();
  for (int idx = tid; idx < 64 * 8; idx += 256) {
    int r = idx >> 3, h = idx & 7;
    wfac[r][h] = ea[(size_t)(e0 + r) * 8 + h] / denom[(size_t)dstL[r] * 8 + h];
  }
  int wave = tid >> 5, lane = tid & 31;
  int rowbase = (lane >> 4) * 8, col0 = lane & 15;

  // gate hidden: tG = silu(elen @ r1 + rb1)   (bf16 in LDS)
#pragma unroll
  for (int t = 0; t < 2; ++t) {
    int tile = wave * 2 + t;
    int rt = tile >> 2, ct = tile & 3;
    v8f acc = {};
    acc = wmma_bf16(fragA(&Eb[rt * 16][0], 64), fragB(r1T + ct * 16 * 64, 64, 0), acc);
    acc = wmma_bf16(fragA(&Eb[rt * 16][32], 64), fragB(r1T + ct * 16 * 64, 64, 32), acc);
    int cg = ct * 16 + col0;
    float rb = rb1[cg];
#pragma unroll
    for (int i = 0; i < 8; ++i)
      tG[rt * 16 + rowbase + i][cg] = f2bf(siluf(acc[i] + rb));
  }
  __syncthreads();   // tG + wfac visible to all waves; no barriers after this

  int eg = wave & 1;        // edge-row group: rows [eg*32, eg*32+32)
  int cq = wave >> 1;       // 128-col quarter
  int r = lane & 15;
  int sb = (lane >> 4) * 8; // s-range for this lane's A-fragment halves
  int R0 = eg * 32 + r, R1 = eg * 32 + 16 + r;

  float sh0[8], sh1[8];
#pragma unroll
  for (int j = 0; j < 8; ++j) {
    sh0[j] = bf2f(shT[R0][sb + j]);
    sh1[j] = bf2f(shT[R1][sb + j]);
  }
  const unsigned* m0p = (const unsigned*)&msgT[R0][0];
  const unsigned* m1p = (const unsigned*)&msgT[R1][0];

  v8f c0[8] = {}, c1[8] = {};
  for (int kb = 0; kb < 32; ++kb) {
    unsigned mp0 = m0p[kb];          // msg[R0][2kb], msg[R0][2kb+1]
    unsigned mp1 = m1p[kb];
    float x0 = bf2f((bf16u)(mp0 & 0xffffu)), y0 = bf2f((bf16u)(mp0 >> 16));
    float x1 = bf2f((bf16u)(mp1 & 0xffffu)), y1 = bf2f((bf16u)(mp1 >> 16));
    FragU a0, a1;
#pragma unroll
    for (int j = 0; j < 8; ++j) {
      a0.h[j]     = f2bf(x0 * sh0[j]);
      a0.h[8 + j] = f2bf(y0 * sh0[j]);
      a1.h[j]     = f2bf(x1 * sh1[j]);
      a1.h[8 + j] = f2bf(y1 * sh1[j]);
    }
#pragma unroll
    for (int nt = 0; nt < 8; ++nt) {
      int colbase = cq * 128 + nt * 16;
      v16bf b = fragB(Wtp2T + (size_t)colbase * 1024, 1024, kb * 32);
      c0[nt] = wmma_bf16(a0.v, b, c0[nt]);   // B fragment reused for both row tiles
      c1[nt] = wmma_bf16(a1.v, b, c1[nt]);
    }
  }

  // epilogue: gate (WMMA), attention weight, scatter-add into node_agg
#pragma unroll
  for (int nt = 0; nt < 8; ++nt) {
    int colbase = cq * 128 + nt * 16;
    int ccol = colbase + col0;
    int h = ccol >> 6;
    float rb = rb2[ccol];
    v8f g0 = {}, g1v = {};
    v16bf bg0 = fragB(r2T + (size_t)colbase * 64, 64, 0);
    v16bf bg1 = fragB(r2T + (size_t)colbase * 64, 64, 32);
    g0  = wmma_bf16(fragA(&tG[eg * 32][0], 64), bg0, g0);
    g0  = wmma_bf16(fragA(&tG[eg * 32][32], 64), bg1, g0);
    g1v = wmma_bf16(fragA(&tG[eg * 32 + 16][0], 64), bg0, g1v);
    g1v = wmma_bf16(fragA(&tG[eg * 32 + 16][32], 64), bg1, g1v);
#pragma unroll
    for (int i = 0; i < 8; ++i) {
      int er0 = eg * 32 + rowbase + i;
      int er1 = eg * 32 + 16 + rowbase + i;
      float v0 = c0[nt][i] * (g0[i] + rb) * wfac[er0][h];
      float v1 = c1[nt][i] * (g1v[i] + rb) * wfac[er1][h];
      atomicAdd(&nagg[(size_t)dstL[er0] * 512 + ccol], v0);
      atomicAdd(&nagg[(size_t)dstL[er1] * 512 + ccol], v1);
    }
  }
}

// ---------------- out = node_agg @ W_lin + b_lin ----------------
__global__ __launch_bounds__(256) void kOut(
    const float* __restrict__ nagg, const bf16u* __restrict__ WlinT,
    const float* __restrict__ b_lin, float* __restrict__ out) {
  __shared__ __align__(16) bf16u Ab[128][32];
  int tid = threadIdx.x;
  int r0 = blockIdx.x * 128;
  int wave = tid >> 5, lane = tid & 31;
  int rowbase = (lane >> 4) * 8, col0 = lane & 15;
  int wr = wave * 16;
  v8f c[4] = {};
  for (int kb = 0; kb < 16; ++kb) {
    __syncthreads();
    for (int idx = tid; idx < 128 * 32; idx += 256) {
      int r = idx >> 5, kk = idx & 31;
      Ab[r][kk] = f2bf(nagg[(size_t)(r0 + r) * 512 + kb * 32 + kk]);
    }
    __syncthreads();
    v16bf a = fragA(&Ab[wr][0], 32);
#pragma unroll
    for (int nt = 0; nt < 4; ++nt) {
      v16bf b = fragB(WlinT + nt * 16 * 512, 512, kb * 32);
      c[nt] = wmma_bf16(a, b, c[nt]);
    }
  }
#pragma unroll
  for (int nt = 0; nt < 4; ++nt) {
    int ccol = nt * 16 + col0;
    float bl = b_lin[ccol];
#pragma unroll
    for (int i = 0; i < 8; ++i)
      out[(size_t)(r0 + wr + rowbase + i) * 64 + ccol] = c[nt][i] + bl;
  }
}

extern "C" void kernel_launch(void* const* d_in, const int* in_sizes, int n_in,
                              void* d_out, int out_size, void* d_ws, size_t ws_size,
                              hipStream_t stream) {
  const float* node_fea = (const float*)d_in[0];
  const float* edge_sh  = (const float*)d_in[1];
  const float* elen     = (const float*)d_in[2];
  const int*   esrc     = (const int*)d_in[3];
  const int*   edst     = (const int*)d_in[4];
  // d_in[5] = batch (unused)
  const float* W_pre = (const float*)d_in[6];
  const float* b_pre = (const float*)d_in[7];
  const float* W_tp2 = (const float*)d_in[8];
  const float* r1    = (const float*)d_in[9];
  const float* rb1   = (const float*)d_in[10];
  const float* r2    = (const float*)d_in[11];
  const float* rb2   = (const float*)d_in[12];
  const float* a1    = (const float*)d_in[13];
  const float* ab1   = (const float*)d_in[14];
  const float* g1    = (const float*)d_in[15];
  const float* be1   = (const float*)d_in[16];
  const float* a2    = (const float*)d_in[17];
  const float* ab2   = (const float*)d_in[18];
  const float* g2    = (const float*)d_in[19];
  const float* be2   = (const float*)d_in[20];
  const float* a3    = (const float*)d_in[21];
  const float* ab3   = (const float*)d_in[22];
  const float* W_lin = (const float*)d_in[23];
  const float* b_lin = (const float*)d_in[24];
  float* out = (float*)d_out;

  const int N = in_sizes[0] / 64;
  const int E = in_sizes[3];

  char* wsp = (char*)d_ws;
  size_t off = 0;
  auto take = [&](size_t bytes) -> char* {
    char* p = wsp + off;
    off += (bytes + 255) & ~(size_t)255;
    return p;
  };
  bf16u* WpreT = (bf16u*)take((size_t)64 * 192 * 2);
  bf16u* r1T   = (bf16u*)take((size_t)64 * 64 * 2);
  bf16u* r2T   = (bf16u*)take((size_t)512 * 64 * 2);
  bf16u* Wtp2T = (bf16u*)take((size_t)512 * 1024 * 2);
  bf16u* WlinT = (bf16u*)take((size_t)64 * 512 * 2);
  bf16u* a1T   = (bf16u*)take((size_t)64 * 64 * 2);
  bf16u* a2T   = (bf16u*)take((size_t)64 * 64 * 2);
  bf16u* a3T   = (bf16u*)take((size_t)16 * 64 * 2);
  bf16u* msgw  = (bf16u*)take((size_t)E * 64 * 2);
  float* eaBuf = (float*)take((size_t)E * 8 * 4);
  float* mbuf  = (float*)take((size_t)N * 8 * 4);
  float* denom = (float*)take((size_t)N * 8 * 4);
  float* nagg  = (float*)take((size_t)N * 512 * 4);
  (void)ws_size; (void)n_in; (void)out_size;

  kConvT<<<48, 256, 0, stream>>>(W_pre, WpreT, 192, 64);
  kConvT<<<16, 256, 0, stream>>>(r1, r1T, 64, 64);
  kConvT<<<128, 256, 0, stream>>>(r2, r2T, 64, 512);
  kConvT<<<2048, 256, 0, stream>>>(W_tp2, Wtp2T, 1024, 512);
  kConvT<<<128, 256, 0, stream>>>(W_lin, WlinT, 512, 64);
  kConvT<<<16, 256, 0, stream>>>(a1, a1T, 64, 64);
  kConvT<<<16, 256, 0, stream>>>(a2, a2T, 64, 64);
  kConvTpad<<<4, 256, 0, stream>>>(a3, a3T, 64, 8, 16);
  kFillF<<<64, 256, 0, stream>>>(mbuf, N * 8, -__builtin_inff());
  kFillF<<<64, 256, 0, stream>>>(denom, N * 8, 0.0f);
  kFillF<<<1024, 256, 0, stream>>>(nagg, N * 512, 0.0f);

  kMsg<<<E / 128, 256, 0, stream>>>(node_fea, elen, esrc, edst, WpreT, b_pre, msgw);
  kAlpha<<<E / 64, 256, 0, stream>>>(elen, edst, a1T, ab1, g1, be1,
                                     a2T, ab2, g2, be2, a3T, ab3, eaBuf, mbuf);
  kEa<<<E / 256, 256, 0, stream>>>(edst, mbuf, eaBuf, denom, E);
  kMain<<<E / 64, 256, 0, stream>>>(msgw, edge_sh, elen, edst, r1T, rb1, r2T, rb2,
                                    Wtp2T, eaBuf, denom, nagg);
  kOut<<<N / 128, 256, 0, stream>>>(nagg, WlinT, b_lin, out);
}
